// Attention_17703855194398
// MI455X (gfx1250) — matrix operand order repaired and checked by reading, not compiled, not run on hardware
//
#include <hip/hip_runtime.h>

// Problem constants (match reference)
#define B_   2
#define L_   2048
#define D_   2048
#define HD_  64
#define NH_  32
#define DC_  2048
#define EPS_ 1.1920929e-07f
// 1/sqrt(64) * log2(e): scores scaled into log2 domain so exp2f suffices
#define SCALE_LOG2 0.18033688011112042f

typedef __attribute__((ext_vector_type(16))) __bf16 v16bf;
typedef __attribute__((ext_vector_type(8)))  float  v8f;

union FragU { v16bf v; uint4 q[2]; unsigned short us[16]; };

__device__ __forceinline__ unsigned short f2bf(float f) {
  unsigned int u = __float_as_uint(f);
  u += 0x7FFFu + ((u >> 16) & 1u);          // round-to-nearest-even
  return (unsigned short)(u >> 16);
}
__device__ __forceinline__ float bf2f(unsigned short h) {
  return __uint_as_float(((unsigned int)h) << 16);
}

// DPP row_xmask:N — butterfly xor within each 16-lane row, pure VALU
// (replaces ds_bpermute-based __shfl_xor for the softmax reductions)
template <int N>
__device__ __forceinline__ float dpp_xor(float x) {
  return __int_as_float(__builtin_amdgcn_update_dpp(
      0, __float_as_int(x), 0x160 | N, 0xf, 0xf, true));
}
__device__ __forceinline__ float row_max16(float v) {
  v = fmaxf(v, dpp_xor<1>(v));
  v = fmaxf(v, dpp_xor<2>(v));
  v = fmaxf(v, dpp_xor<4>(v));
  v = fmaxf(v, dpp_xor<8>(v));
  return v;
}
__device__ __forceinline__ float row_sum16(float v) {
  v += dpp_xor<1>(v);
  v += dpp_xor<2>(v);
  v += dpp_xor<4>(v);
  v += dpp_xor<8>(v);
  return v;
}

// A-operand (16x32, MxK, bf16): lane holds row m=lane&15, K-chunks at
// k0=(lane>>4)*8 and k0+16  (two 16B loads)
__device__ __forceinline__ v16bf load_frag_a(const unsigned short* p, int ld, int kbase) {
  const int lane = threadIdx.x & 31;
  const int row  = lane & 15;
  const int k0   = kbase + ((lane >> 4) << 3);
  const unsigned short* r = p + (size_t)row * ld + k0;
  FragU f;
  f.q[0] = *(const uint4*)(r);
  f.q[1] = *(const uint4*)(r + 16);
  return f.v;
}

// B-operand (32x16, KxN, bf16) sourced from a row-major [N,K] matrix:
// lane holds column n=lane&15, K-chunks at k0=(lane>>4)*8 and k0+16 (two 16B loads)
__device__ __forceinline__ v16bf load_frag_b(const unsigned short* p, int ld, int kbase) {
  const int lane = threadIdx.x & 31;
  const int col  = lane & 15;
  const int k0   = kbase + ((lane >> 4) << 3);
  const unsigned short* r = p + (size_t)col * ld + k0;
  FragU f;
  f.q[0] = *(const uint4*)(r);
  f.q[1] = *(const uint4*)(r + 16);
  return f.v;
}

__device__ __forceinline__ v8f wmma_bf16(v16bf a, v16bf b, v8f c) {
  return __builtin_amdgcn_wmma_f32_16x16x32_bf16(false, a, false, b, (short)0, c,
                                                 false, false);
}

// ---------------- elementwise fp32 -> bf16 ----------------
__global__ __launch_bounds__(256) void k_cvt(const float* __restrict__ src,
                                             unsigned short* __restrict__ dst, int n) {
  int t = blockIdx.x * 256 + threadIdx.x;
  if (t < n) dst[t] = f2bf(src[t]);
}

// ---------------- ss = condition @ w_ada.T  [B, 2D] ----------------
__global__ __launch_bounds__(256) void k_ada(const float* __restrict__ cond,
                                             const float* __restrict__ w_ada,
                                             float* __restrict__ ss) {
  int t = blockIdx.x * 256 + threadIdx.x;          // t < B*2D
  int b = t / (2 * D_);
  int n = t % (2 * D_);
  const float4* c = (const float4*)(cond + (size_t)b * DC_);
  const float4* w = (const float4*)(w_ada + (size_t)n * DC_);
  float acc = 0.f;
  for (int j = 0; j < DC_ / 4; ++j) {
    const float4 cv = c[j];
    const float4 wv = w[j];
    acc = fmaf(cv.x, wv.x, acc);
    acc = fmaf(cv.y, wv.y, acc);
    acc = fmaf(cv.z, wv.z, acc);
    acc = fmaf(cv.w, wv.w, acc);
  }
  ss[t] = acc;
}

// ---------------- h = rmsnorm(x)*(1+scale)+shift -> bf16 ----------------
__global__ __launch_bounds__(256) void k_adarms(const float* __restrict__ x,
                                                const float* __restrict__ ss,
                                                unsigned short* __restrict__ hbf) {
  __shared__ float red[256];
  const int row = blockIdx.x;                      // b*L + l
  const int b   = row / L_;
  const int tid = threadIdx.x;
  const float* xr = x + (size_t)row * D_;
  float xv[8];
  float p = 0.f;
#pragma unroll
  for (int i = 0; i < 8; ++i) { xv[i] = xr[tid + i * 256]; p += xv[i] * xv[i]; }
  red[tid] = p;
  __syncthreads();
  for (int s = 128; s > 0; s >>= 1) {
    if (tid < s) red[tid] += red[tid + s];
    __syncthreads();
  }
  const float rs = rsqrtf(red[0] * (1.0f / D_) + EPS_);
  const float* shift = ss + (size_t)b * (2 * D_);
  const float* scale = shift + D_;
#pragma unroll
  for (int i = 0; i < 8; ++i) {
    const int c = tid + i * 256;
    hbf[(size_t)row * D_ + c] = f2bf(xv[i] * rs * (1.0f + scale[c]) + shift[c]);
  }
}

// ---------------- WMMA GEMM: C[M,N] = A[M,K] @ W[N,K]^T ----------------
// one wave computes a 32x64 tile (2 A-frags reuse each B-frag -> 8 WMMA/step);
// block = 4 waves = 4 tiles
template <bool OUT_BF16>
__global__ __launch_bounds__(128) void k_gemm(const unsigned short* __restrict__ A,
                                              const unsigned short* __restrict__ W,
                                              void* __restrict__ outp,
                                              int M, int N, int K) {
  const int lane = threadIdx.x & 31;
  const int wid  = threadIdx.x >> 5;
  const int tiles_n = N >> 6;
  const int tile  = blockIdx.x * 4 + wid;
  const int tm    = tile / tiles_n;
  const int tn    = tile - tm * tiles_n;
  const int mbase = tm << 5;
  const int nbase = tn << 6;
  const v8f zv = {0.f, 0.f, 0.f, 0.f, 0.f, 0.f, 0.f, 0.f};
  v8f c0[4] = {zv, zv, zv, zv};
  v8f c1[4] = {zv, zv, zv, zv};
  const unsigned short* Ar0 = A + (size_t)mbase * K;
  const unsigned short* Ar1 = A + (size_t)(mbase + 16) * K;
  for (int k = 0; k < K; k += 32) {
    if (k + 32 < K) __builtin_prefetch(Ar0 + (size_t)(lane & 15) * K + k + 32, 0, 1);
    const v16bf a0 = load_frag_a(Ar0, K, k);
    const v16bf a1 = load_frag_a(Ar1, K, k);
#pragma unroll
    for (int f = 0; f < 4; ++f) {
      const v16bf bw = load_frag_b(W + (size_t)(nbase + f * 16) * K, K, k);
      c0[f] = wmma_bf16(a0, bw, c0[f]);
      c1[f] = wmma_bf16(a1, bw, c1[f]);
    }
  }
#pragma unroll
  for (int f = 0; f < 4; ++f) {
#pragma unroll
    for (int r = 0; r < 8; ++r) {
      const int m = mbase + r + ((lane >> 4) << 3);
      const int n = nbase + f * 16 + (lane & 15);
      if (OUT_BF16) {
        ((unsigned short*)outp)[(size_t)m * N + n]        = f2bf(c0[f][r]);
        ((unsigned short*)outp)[(size_t)(m + 16) * N + n] = f2bf(c1[f][r]);
      } else {
        ((float*)outp)[(size_t)m * N + n]        = c0[f][r];
        ((float*)outp)[(size_t)(m + 16) * N + n] = c1[f][r];
      }
    }
  }
}

// ------------- per-head RMSNorm(qk_w) + RoPE on q,k; write bf16 -------------
// one wave per (b, h, l, which); lane owns RoPE pair (d0=2*lane, d1=2*lane+1)
__global__ __launch_bounds__(128) void k_qkpost(const unsigned short* __restrict__ qkv,
                                                const float* __restrict__ rope,
                                                const float* __restrict__ qk_w,
                                                unsigned short* __restrict__ qo,
                                                unsigned short* __restrict__ ko) {
  const int lane = threadIdx.x & 31;
  int idx = blockIdx.x * 4 + (threadIdx.x >> 5);   // < B*NH*L*2
  const int which = idx & 1; idx >>= 1;
  const int l = idx & (L_ - 1); idx >>= 11;
  const int h = idx & (NH_ - 1);
  const int b = idx >> 5;
  const int d0 = lane * 2, d1 = d0 + 1;
  const unsigned short* src =
      qkv + (size_t)(b * L_ + l) * (3 * D_) + (size_t)which * D_ + h * HD_;
  const float x0 = bf2f(src[d0]);
  const float x1 = bf2f(src[d1]);
  float ssq = x0 * x0 + x1 * x1;
  ssq = row_sum16(ssq);                // xor 1/2/4/8 (within 16-lane rows)
  ssq += __shfl_xor(ssq, 16, 32);      // cross-row step
  const float rs = rsqrtf(ssq * (1.0f / HD_) + EPS_);
  const float n0 = x0 * rs * qk_w[d0];
  const float n1 = x1 * rs * qk_w[d1];
  const float* r0 = rope + (size_t)l * HD_;
  const float* r1 = rope + (size_t)L_ * HD_ + (size_t)l * HD_;
  const float o0 = n0 * r0[d0] - n1 * r1[d0];   // x_hat even = -x_odd
  const float o1 = n1 * r0[d1] + n0 * r1[d1];   // x_hat odd  =  x_even
  unsigned short* dst = (which ? ko : qo) + ((size_t)(b * NH_ + h) * L_ + l) * HD_;
  dst[d0] = f2bf(o0);
  dst[d1] = f2bf(o1);
}

// ---------------- V -> V^T per head: vt[b,h,d,l] ----------------
__global__ __launch_bounds__(256) void k_vtrans(const unsigned short* __restrict__ qkv,
                                                unsigned short* __restrict__ vt) {
  const int t = blockIdx.x * 256 + threadIdx.x;   // < B*L*D
  const int d = t & (HD_ - 1);
  const int h = (t >> 6) & (NH_ - 1);
  const int l = (t >> 11) & (L_ - 1);
  const int b = t >> 22;
  vt[((size_t)(b * NH_ + h) * HD_ + d) * L_ + l] =
      qkv[(size_t)(b * L_ + l) * (3 * D_) + 2 * D_ + h * HD_ + d];
}

// ---------------- flash attention: one wave per 16-row Q tile ----------------
__global__ __launch_bounds__(128) void k_flash(const unsigned short* __restrict__ qb,
                                               const unsigned short* __restrict__ kb,
                                               const unsigned short* __restrict__ vt,
                                               unsigned short* __restrict__ ob) {
  __shared__ __align__(16) unsigned short ptile[4][16 * 32];
  const int lane = threadIdx.x & 31;
  const int wid  = threadIdx.x >> 5;
  int tile = blockIdx.x * 4 + wid;                 // < B*NH*(L/16)
  const int qt = tile & 127; tile >>= 7;
  const int h  = tile & (NH_ - 1);
  const int b  = tile >> 5;
  const size_t bh = (size_t)(b * NH_ + h);

  const unsigned short* qp = qb + (bh * L_ + (size_t)qt * 16) * HD_;
  const unsigned short* kp = kb + bh * L_ * HD_;
  const unsigned short* vp = vt + bh * HD_ * L_;

  const v16bf aq0 = load_frag_a(qp, HD_, 0);
  const v16bf aq1 = load_frag_a(qp, HD_, 32);

  const v8f zv = {0.f, 0.f, 0.f, 0.f, 0.f, 0.f, 0.f, 0.f};
  v8f oacc[4] = {zv, zv, zv, zv};
  float mrun[8], lrun[8];
#pragma unroll
  for (int r = 0; r < 8; ++r) { mrun[r] = -1e30f; lrun[r] = 0.f; }

  unsigned short* my = &ptile[wid][0];

  for (int j = 0; j < L_; j += 32) {
    // S = Q K^T (16 x 32 block of scores), f32 accum
    const v16bf bk0a = load_frag_b(kp + (size_t)j * HD_, HD_, 0);
    const v16bf bk0b = load_frag_b(kp + (size_t)j * HD_, HD_, 32);
    const v16bf bk1a = load_frag_b(kp + (size_t)(j + 16) * HD_, HD_, 0);
    const v16bf bk1b = load_frag_b(kp + (size_t)(j + 16) * HD_, HD_, 32);
    v8f s0 = zv, s1 = zv;
    s0 = wmma_bf16(aq0, bk0a, s0);
    s0 = wmma_bf16(aq1, bk0b, s0);
    s1 = wmma_bf16(aq0, bk1a, s1);
    s1 = wmma_bf16(aq1, bk1b, s1);

    // online softmax in log2 domain; row r lives in this lane's 16-lane row,
    // DPP row_xmask butterflies reduce across the 16 columns
    float p0[8], p1[8];
#pragma unroll
    for (int r = 0; r < 8; ++r) {
      const float a  = s0[r] * SCALE_LOG2;
      const float bq = s1[r] * SCALE_LOG2;
      const float mx = row_max16(fmaxf(a, bq));
      const float mnew = fmaxf(mrun[r], mx);
      const float corr = exp2f(mrun[r] - mnew);
      const float e0 = exp2f(a - mnew);
      const float e1 = exp2f(bq - mnew);
      const float rsum = row_sum16(e0 + e1);
      lrun[r] = lrun[r] * corr + rsum;
      mrun[r] = mnew;
      p0[r] = e0;
      p1[r] = e1;
      oacc[0][r] *= corr;
      oacc[1][r] *= corr;
      oacc[2][r] *= corr;
      oacc[3][r] *= corr;
    }

    // C-layout -> A-layout conversion of P via per-wave LDS tile
#pragma unroll
    for (int r = 0; r < 8; ++r) {
      const int m = r + ((lane >> 4) << 3);
      my[m * 32 + (lane & 15)]      = f2bf(p0[r]);
      my[m * 32 + 16 + (lane & 15)] = f2bf(p1[r]);
    }
    __syncthreads();
    const v16bf pa = load_frag_a(my, 32, 0);
    __syncthreads();

    // O += P V  (B-operand from V^T: contiguous along j)
#pragma unroll
    for (int f = 0; f < 4; ++f) {
      const v16bf bv = load_frag_b(vp + (size_t)(f * 16) * L_, L_, j);
      oacc[f] = wmma_bf16(pa, bv, oacc[f]);
    }
  }

  float inv[8];
#pragma unroll
  for (int r = 0; r < 8; ++r) inv[r] = 1.0f / lrun[r];
  unsigned short* op = ob + ((size_t)b * L_ + (size_t)qt * 16) * D_ + h * HD_;
#pragma unroll
  for (int f = 0; f < 4; ++f) {
#pragma unroll
    for (int r = 0; r < 8; ++r) {
      const int m = r + ((lane >> 4) << 3);
      const int n = f * 16 + (lane & 15);
      op[(size_t)m * D_ + n] = f2bf(oacc[f][r] * inv[r]);
    }
  }
}

// ---------------------------------------------------------------------------
static inline size_t align256(size_t x) { return (x + 255) & ~(size_t)255; }

extern "C" void kernel_launch(void* const* d_in, const int* in_sizes, int n_in,
                              void* d_out, int out_size, void* d_ws, size_t ws_size,
                              hipStream_t stream) {
  const float* x      = (const float*)d_in[0];
  const float* cond   = (const float*)d_in[1];
  const float* rope   = (const float*)d_in[2];
  const float* w_ada  = (const float*)d_in[3];
  const float* w_qkv  = (const float*)d_in[4];
  const float* w_out  = (const float*)d_in[5];
  const float* qk_w   = (const float*)d_in[6];
  float* out = (float*)d_out;

  char* ws = (char*)d_ws;
  float* ss = (float*)ws;              ws += align256((size_t)B_ * 2 * D_ * 4);
  unsigned short* wqkv_bf = (unsigned short*)ws; ws += align256((size_t)3 * D_ * D_ * 2);
  unsigned short* wout_bf = (unsigned short*)ws; ws += align256((size_t)D_ * D_ * 2);
  unsigned short* h_bf    = (unsigned short*)ws; ws += align256((size_t)B_ * L_ * D_ * 2);
  unsigned short* qkv_bf  = (unsigned short*)ws; ws += align256((size_t)B_ * L_ * 3 * D_ * 2);
  unsigned short* q_bf    = (unsigned short*)ws; ws += align256((size_t)B_ * L_ * D_ * 2);
  unsigned short* k_bf    = (unsigned short*)ws; ws += align256((size_t)B_ * L_ * D_ * 2);
  unsigned short* vt_bf   = (unsigned short*)ws; ws += align256((size_t)B_ * L_ * D_ * 2);
  unsigned short* o_bf    = (unsigned short*)ws; ws += align256((size_t)B_ * L_ * D_ * 2);

  const int n_wqkv = 3 * D_ * D_;
  const int n_wout = D_ * D_;
  k_cvt<<<(n_wqkv + 255) / 256, 256, 0, stream>>>(w_qkv, wqkv_bf, n_wqkv);
  k_cvt<<<(n_wout + 255) / 256, 256, 0, stream>>>(w_out, wout_bf, n_wout);

  k_ada<<<(B_ * 2 * D_) / 256, 256, 0, stream>>>(cond, w_ada, ss);
  k_adarms<<<B_ * L_, 256, 0, stream>>>(x, ss, h_bf);

  // QKV: [B*L, 3D] = h @ w_qkv^T   (32x64 tiles, 4 waves/block)
  k_gemm<true><<<(B_ * L_ / 32) * (3 * D_ / 64) / 4, 128, 0, stream>>>(
      h_bf, wqkv_bf, (void*)qkv_bf, B_ * L_, 3 * D_, D_);

  k_qkpost<<<(B_ * NH_ * L_ * 2) / 4, 128, 0, stream>>>(qkv_bf, rope, qk_w, q_bf, k_bf);
  k_vtrans<<<(B_ * L_ * D_) / 256, 256, 0, stream>>>(qkv_bf, vt_bf);

  // SDPA: one wave per 16-row Q tile, 4 waves/block
  k_flash<<<(B_ * NH_ * (L_ / 16)) / 4, 128, 0, stream>>>(q_bf, k_bf, vt_bf, o_bf);

  // out = o @ w_out^T  (fp32 result)
  k_gemm<false><<<(B_ * L_ / 32) * (D_ / 64) / 4, 128, 0, stream>>>(
      o_bf, wout_bf, (void*)out, B_ * L_, D_, D_);
}
